// MultiHeadedAttention_86466281603674
// MI455X (gfx1250) — matrix operand order repaired and checked by reading, not compiled, not run on hardware
//
#include <hip/hip_runtime.h>
#include <math.h>

// ---------------- problem constants ----------------
#define Bb 2
#define Ss 2048
#define Dd 1024
#define Hh 16
#define DKk 64
#define BSD (Bb * Ss * Dd)   // 4,194,304
#define DDq (Dd * Dd)        // 1,048,576
#define CHUNK 64             // keys staged per LDS buffer

// ---------------- WMMA vector types ----------------
typedef __attribute__((ext_vector_type(16))) __bf16 v16bf;
typedef __attribute__((ext_vector_type(8)))  float  v8f;
typedef __attribute__((ext_vector_type(4)))  unsigned int uint4v;

// fp32 -> bf16 (round to nearest even), stored as raw u16
__device__ __forceinline__ unsigned short f32_to_bf16(float f) {
    unsigned int u = __float_as_uint(f);
    unsigned int r = (u + 0x7FFFu + ((u >> 16) & 1u)) >> 16;
    return (unsigned short)r;
}

// A-fragment: per lane, 16 contiguous K values starting at
// (lane>=16 ? 16 : 0) (same k order as B). `p` points at that run.
__device__ __forceinline__ v16bf load_frag_a(const unsigned short* p) {
    union { uint4v u[2]; v16bf v; } cvt;
    cvt.u[0] = *(const uint4v*)(p);        // elements 0..7
    cvt.u[1] = *(const uint4v*)(p + 8);    // elements 8..15
    return cvt.v;
}

// B-fragment (same k order as A): per lane, 16 contiguous K values
// starting at (lane>=16 ? 16 : 0). `p` points at that run.
__device__ __forceinline__ v16bf load_frag_b(const unsigned short* p) {
    union { uint4v u[2]; v16bf v; } cvt;
    cvt.u[0] = *(const uint4v*)(p);
    cvt.u[1] = *(const uint4v*)(p + 8);
    return cvt.v;
}

__device__ __forceinline__ v8f wmma_bf16(v16bf a, v16bf b, v8f c) {
    return __builtin_amdgcn_wmma_f32_16x16x32_bf16(
        false, a, false, b, (short)0, c, false, false);
}

// ---------------- async global -> LDS (ASYNCcnt-tracked) ----------------
// 16-byte per-lane async copy; LDS dest given as wave-relative byte offset.
__device__ __forceinline__ void async_cp_b128(unsigned lds_byte,
                                              const unsigned short* g) {
    unsigned long long ga = (unsigned long long)(uintptr_t)g;
    asm volatile("global_load_async_to_lds_b128 %0, %1, off"
                 :: "v"(lds_byte), "v"(ga) : "memory");
}

#if defined(__has_builtin) && __has_builtin(__builtin_amdgcn_s_wait_asynccnt)
#define WAIT_ASYNCCNT(n) __builtin_amdgcn_s_wait_asynccnt(n)
#else
#define WAIT_ASYNCCNT(n) asm volatile("s_wait_asynccnt " #n ::: "memory")
#endif

// ---------------- prep: fp32 -> bf16 copy ----------------
__global__ __launch_bounds__(256) void cvt_bf16_kernel(
    const float* __restrict__ in, unsigned short* __restrict__ out, int n) {
    int i = blockIdx.x * 256 + threadIdx.x;
    if (i < n) out[i] = f32_to_bf16(in[i]);
}

// ---------------- prep: fp32 [K][N] -> bf16 [N][K] (transpose) ----------------
__global__ __launch_bounds__(256) void transpose_cvt_kernel(
    const float* __restrict__ in, unsigned short* __restrict__ out) {
    int i = blockIdx.x * 256 + threadIdx.x;       // over Dd*Dd
    int k = i >> 10;
    int n = i & (Dd - 1);
    out[(size_t)n * Dd + k] = f32_to_bf16(in[i]);
}

// ---------------- projection GEMM: Out[M,N] = X[M,K] @ W + bias (bf16 out) ----
// X bf16 row-major [4096,1024]; WT bf16 [N][K] (transposed weights);
// wave => 16 rows x 64 cols (4 accumulators, A reused 4x), 8 waves/block.
__global__ __launch_bounds__(256) void proj_gemm_kernel(
    const unsigned short* __restrict__ X,
    const unsigned short* __restrict__ WT,
    const float* __restrict__ bias,
    unsigned short* __restrict__ Out) {
    const int N = Dd, K = Dd;
    int lane = threadIdx.x & 31;
    int wave = threadIdx.x >> 5;
    int bm = blockIdx.x >> 3;                 // M tiles of 64   (64 blocks)
    int bn = blockIdx.x & 7;                  // N tiles of 128  (8 blocks)
    int row_base = bm * 64 + (wave & 3) * 16;
    int col_base = bn * 128 + (wave >> 2) * 64;
    int l15   = lane & 15;
    int khalf = (lane >> 4) << 3;             // 0 or 8  (A pattern)
    int k16   = (lane >> 4) << 4;             // 0 or 16 (B pattern)

    const unsigned short* Arow = X + (size_t)(row_base + l15) * K + k16;
    const unsigned short* Br0  = WT + (size_t)(col_base +  0 + l15) * K + k16;
    const unsigned short* Br1  = WT + (size_t)(col_base + 16 + l15) * K + k16;
    const unsigned short* Br2  = WT + (size_t)(col_base + 32 + l15) * K + k16;
    const unsigned short* Br3  = WT + (size_t)(col_base + 48 + l15) * K + k16;

    v8f acc0 = {}, acc1 = {}, acc2 = {}, acc3 = {};
    for (int kb = 0; kb < K; kb += 32) {
        v16bf a  = load_frag_a(Arow + kb);
        v16bf b0 = load_frag_b(Br0 + kb);
        v16bf b1 = load_frag_b(Br1 + kb);
        v16bf b2 = load_frag_b(Br2 + kb);
        v16bf b3 = load_frag_b(Br3 + kb);
        acc0 = wmma_bf16(a, b0, acc0);
        acc1 = wmma_bf16(a, b1, acc1);
        acc2 = wmma_bf16(a, b2, acc2);
        acc3 = wmma_bf16(a, b3, acc3);
    }

    // epilogue: bias add, bf16 store. C layout: VGPR r, lane l ->
    // row = r + 8*(l>=16), col = l&15
    int rhalf = (lane >> 4) << 3;
    float bv0 = bias[col_base +  0 + l15];
    float bv1 = bias[col_base + 16 + l15];
    float bv2 = bias[col_base + 32 + l15];
    float bv3 = bias[col_base + 48 + l15];
#pragma unroll
    for (int r = 0; r < 8; ++r) {
        size_t row = (size_t)(row_base + r + rhalf) * N;
        Out[row + col_base +  0 + l15] = f32_to_bf16(acc0[r] + bv0);
        Out[row + col_base + 16 + l15] = f32_to_bf16(acc1[r] + bv1);
        Out[row + col_base + 32 + l15] = f32_to_bf16(acc2[r] + bv2);
        Out[row + col_base + 48 + l15] = f32_to_bf16(acc3[r] + bv3);
    }
}

// ---------------- fused attention (flash-style, LDS-staged, double-buffered) --
// All 8 waves of a block share (b,h); K chunks (64 keys x 64 d = 8KB) are
// DMA'd once per block into LDS via the async engine (ASYNCcnt) while the
// previous chunk's 8 WMMAs + exp accumulation execute.
__global__ __launch_bounds__(256) void attn_kernel(
    const unsigned short* __restrict__ Qp,
    const unsigned short* __restrict__ Kp,
    const float* __restrict__ value,
    float* __restrict__ xh) {            // xh: [B,H,S] fp32
    __shared__ unsigned short ktile[2][CHUNK * DKk];   // 2 x 8KB

    int tid  = threadIdx.x;
    int lane = tid & 31;
    int wave = tid >> 5;
    int gt = blockIdx.x * 8 + wave;      // 0..4095 query tiles
    int qt = gt & 127;                   // S/16 = 128 tiles; 8 waves same b,h
    int h  = (gt >> 7) & 15;
    int b  = gt >> 11;
    int q_base = qt * 16;
    int l15   = lane & 15;
    int khalf = (lane >> 4) << 3;
    int k16   = (lane >> 4) << 4;

    // A fragments: 16 queries x DK=64 -> two k-steps, loaded once.
    const unsigned short* qrow =
        Qp + (size_t)(b * Ss + q_base + l15) * Dd + h * DKk + k16;
    v16bf A0 = load_frag_a(qrow);
    v16bf A1 = load_frag_a(qrow + 32);

    // staging: thread t copies 32B of the 8KB chunk: row = t/4, elem = (t%4)*16
    const unsigned short* Ksrc = Kp + (size_t)b * Ss * Dd + h * DKk;
    int srow = tid >> 2;
    int seoff = (tid & 3) << 4;
    unsigned lbyte0 = (unsigned)(uintptr_t)&ktile[0][srow * DKk + seoff];
    unsigned lbyte1 = (unsigned)(uintptr_t)&ktile[1][srow * DKk + seoff];

    const float* vbase = value + b * Ss + l15;

    float lsum[8], wsum[8];
#pragma unroll
    for (int r = 0; r < 8; ++r) { lsum[r] = 0.f; wsum[r] = 0.f; }

    // prologue: DMA chunks 0 and 1
    {
        const unsigned short* g0 = Ksrc + (size_t)(0 * CHUNK + srow) * Dd + seoff;
        async_cp_b128(lbyte0,      g0);
        async_cp_b128(lbyte0 + 16, g0 + 8);
        const unsigned short* g1 = Ksrc + (size_t)(1 * CHUNK + srow) * Dd + seoff;
        async_cp_b128(lbyte1,      g1);
        async_cp_b128(lbyte1 + 16, g1 + 8);
    }

    const int nchunks = Ss / CHUNK;      // 32
    for (int ci = 0; ci < nchunks; ++ci) {
        if (ci < nchunks - 1) { WAIT_ASYNCCNT(2); }  // own chunk ci landed
        else                  { WAIT_ASYNCCNT(0); }
        __syncthreads();                 // all waves' DMAs for ci landed

        const unsigned short* tbuf = ktile[ci & 1];
        int kb = ci * CHUNK;
#pragma unroll
        for (int c = 0; c < 4; ++c) {    // 4 column tiles of 16 keys
            const unsigned short* bp = tbuf + (c * 16 + l15) * DKk + k16;
            v16bf B0 = load_frag_b(bp);          // ds_load_b128 x2
            v16bf B1 = load_frag_b(bp + 32);
            v8f acc = {};
            acc = wmma_bf16(A0, B0, acc);
            acc = wmma_bf16(A1, B1, acc);
            float vv = vbase[kb + c * 16];       // this lane's key value
#pragma unroll
            for (int r = 0; r < 8; ++r) {
                float e = __expf(acc[r] * 0.125f);   // 1/sqrt(64)
                lsum[r] += e;
                wsum[r] += e * vv;
            }
        }
        __syncthreads();                 // buffer free to overwrite

        if (ci + 2 < nchunks) {          // DMA chunk ci+2 into buffer (ci&1)
            const unsigned short* g =
                Ksrc + (size_t)((ci + 2) * CHUNK + srow) * Dd + seoff;
            unsigned lb = (ci & 1) ? lbyte1 : lbyte0;
            async_cp_b128(lb,      g);
            async_cp_b128(lb + 16, g + 8);
        }
    }

    // reduce across the 16 key-columns (lanes within each half-wave)
#pragma unroll
    for (int r = 0; r < 8; ++r) {
#pragma unroll
        for (int m = 1; m < 16; m <<= 1) {
            lsum[r] += __shfl_xor(lsum[r], m, 32);
            wsum[r] += __shfl_xor(wsum[r], m, 32);
        }
    }

    if (l15 == 0) {
        int rbase = q_base + ((lane >> 4) << 3);   // rows 0..7 / 8..15
        float* dst = xh + (size_t)(b * Hh + h) * Ss + rbase;
#pragma unroll
        for (int r = 0; r < 8; ++r) dst[r] = wsum[r] / lsum[r];
    }
}

// ---------------- mean over heads -> d_out [B,S,1] ----------------
__global__ __launch_bounds__(256) void reduce_heads_kernel(
    const float* __restrict__ xh, float* __restrict__ out) {
    int i = blockIdx.x * 256 + threadIdx.x;    // 0..B*S-1
    if (i < Bb * Ss) {
        int b = i >> 11;
        int q = i & (Ss - 1);
        float s = 0.f;
#pragma unroll
        for (int h = 0; h < Hh; ++h) s += xh[(size_t)(b * Hh + h) * Ss + q];
        out[i] = s * (1.0f / Hh);
    }
}

extern "C" void kernel_launch(void* const* d_in, const int* in_sizes, int n_in,
                              void* d_out, int out_size, void* d_ws, size_t ws_size,
                              hipStream_t stream) {
    (void)in_sizes; (void)n_in; (void)out_size; (void)ws_size;
    const float* query = (const float*)d_in[0];
    const float* key   = (const float*)d_in[1];
    const float* value = (const float*)d_in[2];
    const float* Wq    = (const float*)d_in[3];
    const float* bq    = (const float*)d_in[4];
    const float* Wk    = (const float*)d_in[5];
    const float* bk    = (const float*)d_in[6];
    float* out = (float*)d_out;

    // workspace layout (bf16 stored as u16)
    unsigned short* Xq  = (unsigned short*)d_ws;     // B*S*D
    unsigned short* Xk  = Xq  + BSD;                 // B*S*D
    unsigned short* WqT = Xk  + BSD;                 // D*D
    unsigned short* WkT = WqT + DDq;                 // D*D
    unsigned short* Qp  = WkT + DDq;                 // B*S*D
    unsigned short* Kp  = Qp  + BSD;                 // B*S*D
    float*          xh  = (float*)(Kp + BSD);        // B*H*S

    cvt_bf16_kernel<<<BSD / 256, 256, 0, stream>>>(query, Xq, BSD);
    cvt_bf16_kernel<<<BSD / 256, 256, 0, stream>>>(key,   Xk, BSD);
    transpose_cvt_kernel<<<DDq / 256, 256, 0, stream>>>(Wq, WqT);
    transpose_cvt_kernel<<<DDq / 256, 256, 0, stream>>>(Wk, WkT);

    // M=4096, wave tile 16x64, block tile 64x128 -> (4096/64)*(1024/128)=512
    proj_gemm_kernel<<<512, 256, 0, stream>>>(Xq, WqT, bq, Qp);
    proj_gemm_kernel<<<512, 256, 0, stream>>>(Xk, WkT, bk, Kp);

    // 4096 query tiles / 8 waves per block
    attn_kernel<<<512, 256, 0, stream>>>(Qp, Kp, value, xh);

    reduce_heads_kernel<<<(Bb * Ss) / 256, 256, 0, stream>>>(xh, out);
}
